// FullDualColumn_33337536152376
// MI455X (gfx1250) — matrix operand order repaired and checked by reading, not compiled, not run on hardware
//
#include <hip/hip_runtime.h>

// ---------------------------------------------------------------------------
// MI455X (gfx1250) SNN layer:
//   conv-as-GEMM via v_wmma_f32_16x16x32_f16 (M=32*192, N=1280, K=48*800),
//   register-blocked 2 t-tiles x 5 o-tiles per wave, then per-(b,n) WTA scan.
// ---------------------------------------------------------------------------

typedef __attribute__((ext_vector_type(16))) _Float16 v16h;
typedef __attribute__((ext_vector_type(8)))  _Float16 v8h;
typedef __attribute__((ext_vector_type(8)))  float    v8f;

#define SNN_B     32
#define SNN_T     128
#define SNN_S     784
#define SNN_SP    800            // synapses padded to 25 chunks of 32
#define SNN_O     1280           // C_OUT * NEUR
#define SNN_C     10
#define SNN_N     128
#define SNN_KW    48             // temporal kernel width
#define SNN_TOUT  177            // T + KERNEL + 1
#define SNN_TTILE 12             // ceil(177/16)
#define SNN_TP    192            // padded pot time dim (12 tiles * 16)
#define SNN_TPAD  240            // padded xT time dim (max index 238)
#define SNN_SCH   25             // s-chunks of 32
#define SNN_CHUNKS (SNN_KW * SNN_SCH)   // 1200 K-chunks of 32
#define SNN_OT    (SNN_O / 16)          // 80 o-tiles
#define SNN_JT    5              // o-tiles per wave (8 waves * 5 * 2 groups = 80)
#define SNN_MT    2              // t-tiles per wave (6 t-pairs cover 12 tiles)
#define SNN_THETA 235.2f
#define SNN_BIAS_THETA 117.6f    // BIAS * THETA
#define SNN_FODEP 48

// triangular step-fire-leak kernel value at integer lag m (un-flipped form;
// the conv derivation absorbed the flip + padding into x[t-1-m]).
__device__ __forceinline__ float snn_kfun(float w, float m) {
    return fmaxf(0.0f, fminf(m * (1.0f / 16.0f), 1.5f * w - m * (1.0f / 32.0f)));
}

// --------------------------------------------------------------------------
// xT[b][tp][s] (f16, zero padded): tp = t + 48, real data tp in [48,176), s<784
// --------------------------------------------------------------------------
__global__ void snn_prep_xt(const float* __restrict__ x, _Float16* __restrict__ xT) {
    int idx = blockIdx.x * blockDim.x + threadIdx.x;
    if (idx >= SNN_B * SNN_TPAD * SNN_SP) return;
    int s  = idx % SNN_SP;
    int r  = idx / SNN_SP;
    int tp = r % SNN_TPAD;
    int b  = r / SNN_TPAD;
    int t  = tp - 48;
    float v = 0.0f;
    if (t >= 0 && t < SNN_T && s < SNN_S)
        v = x[((size_t)b * SNN_S + s) * SNN_T + t];
    xT[idx] = (_Float16)v;
}

// --------------------------------------------------------------------------
// B fragments pre-swizzled into WMMA lane order.
//   bfrag[((chunk*80 + ot)*32 + lane)*16 + j] = kern(o = ot*16 + (lane&15),
//            s = s0 + 8*(lane>>4) + (j&7) + 16*(j>>3), m = chunk/25),  s0 = (chunk%25)*32
// (B 32x16 f16 layout: lanes 0-15 hold K=0..7,16..23, lanes 16-31 hold K=8..15,24..31,
//  two K per VGPR — the ISA's K-striping for 16-bit B operands.)
// --------------------------------------------------------------------------
__global__ void snn_prep_bfrag(const float* __restrict__ wp,
                               const float* __restrict__ wn,
                               _Float16* __restrict__ bfrag) {
    int tid = blockIdx.x * blockDim.x + threadIdx.x;
    if (tid >= SNN_CHUNKS * SNN_OT * 32) return;
    int lane  = tid & 31;
    int r     = tid >> 5;
    int ot    = r % SNN_OT;
    int chunk = r / SNN_OT;
    int m     = chunk / SNN_SCH;
    int s0    = (chunk % SNN_SCH) * 32;
    int half  = lane >> 4;
    int o     = ot * 16 + (lane & 15);
    float fm  = (float)m;
    v16h out;
    #pragma unroll
    for (int j = 0; j < 16; ++j) {
        int s = s0 + 8 * half + (j & 7) + 16 * (j >> 3);
        float v = 0.0f;
        if (s < SNN_S) {
            float p = wp[(size_t)o * SNN_S + s];
            float n = wn[(size_t)o * SNN_S + s];
            v = snn_kfun(p, fm) - 0.05f * snn_kfun(n, fm);
        }
        out[j] = (_Float16)v;
    }
    *(v16h*)(bfrag + (size_t)tid * 16) = out;
}

// per-output-channel bias: 0.05 * mean(w_neg[o,:]) + BIAS*THETA
__global__ void snn_prep_bias(const float* __restrict__ wn, float* __restrict__ bias) {
    int o = blockIdx.x * blockDim.x + threadIdx.x;
    if (o >= SNN_O) return;
    float s = 0.0f;
    for (int i = 0; i < SNN_S; ++i) s += wn[(size_t)o * SNN_S + i];
    bias[o] = 0.05f * (s * (1.0f / SNN_S)) + SNN_BIAS_THETA;
}

// --------------------------------------------------------------------------
// Conv-as-GEMM. Block = 256 threads = 8 waves; block owns (b, t-pair, o-group).
// Wave w owns 2 t-tiles (t0, t0+16) x 5 consecutive o-tiles: each B fragment
// is loaded once per K-chunk and feeds 2 WMMAs; each A fragment feeds 5.
// Inner body: 2 A-loads + 5 B-loads -> 10 WMMAs.
//   A 16x32 f16: lane M = lane&15, VGPR0-3: K = 8*half+{0..7},
//                VGPR4-7: K = 16+8*half+{0..7}  -> two b128 loads/lane.
// --------------------------------------------------------------------------
__global__ void __launch_bounds__(256)
snn_conv_wmma(const _Float16* __restrict__ xT,
              const _Float16* __restrict__ bfrag,
              const float* __restrict__ bias,
              float* __restrict__ pot) {
    int block = blockIdx.x;
    int b  = block % SNN_B;
    int tp = (block / SNN_B) % (SNN_TTILE / SNN_MT);
    int og = block / (SNN_B * (SNN_TTILE / SNN_MT));
    int t0 = tp * (16 * SNN_MT);

    int wave = threadIdx.x >> 5;
    int lane = threadIdx.x & 31;
    int half = lane >> 4;
    int row  = lane & 15;               // A-matrix M index (output time row)
    int ot0  = og * (8 * SNN_JT) + wave * SNN_JT;

    const _Float16* xbase = xT + (size_t)b * SNN_TPAD * SNN_SP;
    // m = 0 row pointers: tp_row = t0 + i*16 + row - 1 - m + 48; -1 row per m
    const _Float16* arow[SNN_MT];
    #pragma unroll
    for (int i = 0; i < SNN_MT; ++i)
        arow[i] = xbase + (size_t)(t0 + i * 16 + row + 47) * SNN_SP + 8 * half;

    const _Float16* bptr  = bfrag + ((size_t)ot0 * 32 + lane) * 16;
    const size_t bstep = (size_t)SNN_OT * 32 * 16;   // halves per K-chunk
    const size_t btile = (size_t)32 * 16;            // halves per o-tile

    v8f acc[SNN_MT][SNN_JT];
    #pragma unroll
    for (int i = 0; i < SNN_MT; ++i)
        #pragma unroll
        for (int j = 0; j < SNN_JT; ++j) acc[i][j] = (v8f){};

    union AV { v16h v; v8h h[2]; };

    for (int m = 0; m < SNN_KW; ++m) {
        size_t aoff = 0;
        #pragma unroll 2
        for (int sc = 0; sc < SNN_SCH; ++sc) {
            __builtin_prefetch(bptr + bstep, 0, 1);   // global_prefetch_b8: next chunk
            AV a[SNN_MT];
            #pragma unroll
            for (int i = 0; i < SNN_MT; ++i) {
                a[i].h[0] = *(const v8h*)(arow[i] + aoff);
                a[i].h[1] = *(const v8h*)(arow[i] + aoff + 16);
            }
            #pragma unroll
            for (int j = 0; j < SNN_JT; ++j) {
                v16h bv = *(const v16h*)(bptr + (size_t)j * btile);
                #pragma unroll
                for (int i = 0; i < SNN_MT; ++i)
                    acc[i][j] = __builtin_amdgcn_wmma_f32_16x16x32_f16(
                        false, a[i].v, false, bv, (short)0, acc[i][j], false, false);
            }
            aoff += 32;
            bptr += bstep;
        }
        #pragma unroll
        for (int i = 0; i < SNN_MT; ++i) arow[i] -= SNN_SP;  // next lag
    }

    // D 16x16 f32: lane N = lane&15 (= o), VGPR v: M = v + 8*half (= t row)
    #pragma unroll
    for (int j = 0; j < SNN_JT; ++j) {
        int o = (ot0 + j) * 16 + (lane & 15);
        float bo = bias[o];
        #pragma unroll
        for (int i = 0; i < SNN_MT; ++i) {
            #pragma unroll
            for (int v = 0; v < 8; ++v) {
                int t = t0 + i * 16 + v + 8 * half;   // t < 192; pot padded
                pot[((size_t)b * SNN_TP + t) * SNN_O + o] = acc[i][j][v] + bo;
            }
        }
    }
}

// --------------------------------------------------------------------------
// Winner-take-all with forced depression: one thread per (b, n) sequence.
// dep is uniform across channels in the reference (broadcast add), so scalar.
// --------------------------------------------------------------------------
__global__ void snn_wta(const float* __restrict__ pot, float* __restrict__ out) {
    int tid = blockIdx.x * blockDim.x + threadIdx.x;
    if (tid >= SNN_B * SNN_N) return;
    int n = tid % SNN_N;
    int b = tid / SNN_N;
    int dep = 0;
    for (int t = 0; t < SNN_TOUT; ++t) {
        const float* p = pot + ((size_t)b * SNN_TP + t) * SNN_O + n;
        int winner = 0;
        int spike = 0;
        if (dep == 0) {
            float best = p[0];
            #pragma unroll
            for (int c = 1; c < SNN_C; ++c) {
                float v = p[(size_t)c * SNN_N];
                if (v > best) { best = v; winner = c; }   // first-max tie rule
            }
            spike = best > SNN_THETA;
        }
        #pragma unroll
        for (int c = 0; c < SNN_C; ++c)
            out[(((size_t)b * SNN_C + c) * SNN_N + n) * SNN_TOUT + t] =
                (spike && c == winner) ? 1.0f : 0.0f;
        dep += spike ? SNN_FODEP : 0;
        dep -= 1;
        if (dep < 0) dep = 0;           // upper clamp to 47 is implicit (48-1)
    }
}

// --------------------------------------------------------------------------
// Workspace layout (bytes, all 256-aligned):
//   xT    12,288,000   bias 5,120(+pad)   bfrag 98,304,000   pot 31,457,280
//   total ~142.1 MB
// --------------------------------------------------------------------------
extern "C" void kernel_launch(void* const* d_in, const int* in_sizes, int n_in,
                              void* d_out, int out_size, void* d_ws, size_t ws_size,
                              hipStream_t stream) {
    const float* x  = (const float*)d_in[0];   // input_spikes (32,1,784,128)
    const float* wp = (const float*)d_in[1];   // weight_pos   (1280,784)
    const float* wn = (const float*)d_in[2];   // weight_neg   (1280,784)
    float* out = (float*)d_out;                // (32,10,128,177)

    char* ws = (char*)d_ws;
    const size_t off_xt    = 0;
    const size_t off_bias  = off_xt   + (size_t)SNN_B * SNN_TPAD * SNN_SP * 2;      // 12,288,000
    const size_t off_bfrag = off_bias + (size_t)SNN_O * 4 + 3840;                   // -> 12,296,960
    const size_t off_pot   = off_bfrag + (size_t)SNN_CHUNKS * SNN_OT * 32 * 16 * 2; // -> 110,600,960

    _Float16* xT    = (_Float16*)(ws + off_xt);
    float*    bias  = (float*)   (ws + off_bias);
    _Float16* bfrag = (_Float16*)(ws + off_bfrag);
    float*    pot   = (float*)   (ws + off_pot);

    { // transpose + pad spikes to f16
        int n = SNN_B * SNN_TPAD * SNN_SP;            // 6,144,000
        snn_prep_xt<<<(n + 255) / 256, 256, 0, stream>>>(x, xT);
    }
    { // build pre-swizzled f16 B fragments
        int n = SNN_CHUNKS * SNN_OT * 32;             // 3,072,000
        snn_prep_bfrag<<<(n + 255) / 256, 256, 0, stream>>>(wp, wn, bfrag);
    }
    snn_prep_bias<<<(SNN_O + 255) / 256, 256, 0, stream>>>(wn, bias);

    { // WMMA GEMM: 32 b * 6 t-pairs * 2 o-groups = 384 blocks, 8 waves each
        int blocks = SNN_B * (SNN_TTILE / SNN_MT) * 2;
        snn_conv_wmma<<<blocks, 256, 0, stream>>>(xT, bfrag, bias, pot);
    }
    { // WTA scan: 4096 sequences
        int n = SNN_B * SNN_N;
        snn_wta<<<(n + 255) / 256, 256, 0, stream>>>(pot, out);
    }
    (void)in_sizes; (void)n_in; (void)out_size; (void)ws_size;
}
